// SparseAttention_2954937500304
// MI455X (gfx1250) — matrix operand order repaired and checked by reading, not compiled, not run on hardware
//
#include <hip/hip_runtime.h>
#include <hip/hip_bf16.h>

typedef __bf16 bf16_t;
typedef bf16_t v16bf __attribute__((ext_vector_type(16)));
typedef bf16_t v8bf  __attribute__((ext_vector_type(8)));
typedef bf16_t v4bf  __attribute__((ext_vector_type(4)));
typedef float  v8f   __attribute__((ext_vector_type(8)));
typedef float  v4f   __attribute__((ext_vector_type(4)));

#define DIM   2048
#define SEQ   2048
#define BATCH 2
#define HEADS 32
#define HD    64
#define WIN   128
#define MROWS (BATCH*SEQ)   /* 4096 */
#define KT    320           /* keys covered per 64-query block: 64 + 2*128 */

__device__ __forceinline__ v8f zero_v8f() {
  v8f v;
#pragma unroll
  for (int i = 0; i < 8; i++) v[i] = 0.f;
  return v;
}

// A-matrix fragment (16x32 bf16): lane holds row (lane%16); element e -> K = 8h+e (e<8), 16+8h+(e-8) (e>=8)
__device__ __forceinline__ v16bf frag_a(const bf16_t* row, int h) {
  v8bf lo = *(const v8bf*)(row + 8 * h);
  v8bf hi = *(const v8bf*)(row + 16 + 8 * h);
  return __builtin_shufflevector(lo, hi, 0,1,2,3,4,5,6,7,8,9,10,11,12,13,14,15);
}
// B-matrix fragment (32x16 bf16): lane holds col (lane%16); element e -> K = 8h+e (e<8), 16+8h+(e-8) (e>=8)
__device__ __forceinline__ v16bf frag_b(const bf16_t* row, int h) {
  v8bf lo = *(const v8bf*)(row + 8 * h);
  v8bf hi = *(const v8bf*)(row + 16 + 8 * h);
  return __builtin_shufflevector(lo, hi, 0,1,2,3,4,5,6,7,8,9,10,11,12,13,14,15);
}
__device__ __forceinline__ v8f wmma_bf16(v16bf a, v16bf b, v8f c) {
  return __builtin_amdgcn_wmma_f32_16x16x32_bf16(false, a, false, b, (short)0, c, false, false);
}

// ---- CDNA5 async global->LDS copy (16B per lane), tracked by ASYNCcnt -----
// Generic pointers to LDS carry the LDS byte offset in their low 32 bits
// (aperture addressing), which is exactly the VDST operand of the async op.
__device__ __forceinline__ void async_ld16(void* lds_dst, const void* gsrc) {
  unsigned lds_off = (unsigned)(uintptr_t)lds_dst;
  unsigned long long ga = (unsigned long long)(uintptr_t)gsrc;
  asm volatile("global_load_async_to_lds_b128 %0, %1, off"
               :: "v"(lds_off), "v"(ga) : "memory");
}
__device__ __forceinline__ void wait_async0() {
  asm volatile("s_wait_asynccnt 0x0" ::: "memory");
}

// ---------------------------------------------------------------------------
// Tiled GEMM: Out[m][n] = sum_k A[m][k] * W[n][k]   (M=4096, N=2048, K=2048)
// A_BF16: 0 -> A is f32 (converted to bf16 on LDS fill), 1 -> A is bf16
// OMODE : 0 -> bf16 out [B,H,S,Hd] (Q/K), 1 -> bf16 out [B,H,Hd,S] (V^T),
//         2 -> f32 out row-major [M][N] (final projection)
// ---------------------------------------------------------------------------
template <int A_BF16, int OMODE>
__global__ __launch_bounds__(256) void gemm128(const void* __restrict__ Ap,
                                               const float* __restrict__ W,
                                               void* __restrict__ Out) {
  __shared__ alignas(32) bf16_t Abuf[2][128 * 32];
  __shared__ alignas(32) bf16_t Bbuf[2][128 * 32];

  const int t    = threadIdx.x;
  const int wave = t >> 5, lane = t & 31;
  const int m0 = blockIdx.y * 128;
  const int n0 = blockIdx.x * 128;
  const int wm = (wave >> 1) * 32;   // 4 wave-rows
  const int wn = (wave & 1) * 64;    // 2 wave-cols
  const int lm = lane & 15, h = lane >> 4;

  v8f acc[2][4];
#pragma unroll
  for (int i = 0; i < 2; i++)
#pragma unroll
    for (int j = 0; j < 4; j++) acc[i][j] = zero_v8f();

  auto loadA = [&](int kk, int buf) {
    if (A_BF16) {
      const bf16_t* A = (const bf16_t*)Ap;
#pragma unroll
      for (int it = 0; it < 2; it++) {
        int idx = it * 256 + t;             // 512 chunks of 8 bf16
        int row = idx >> 2, c8 = idx & 3;
        v8bf v = *(const v8bf*)(A + (size_t)(m0 + row) * DIM + kk + c8 * 8);
        *(v8bf*)&Abuf[buf][row * 32 + c8 * 8] = v;
      }
    } else {
      const float* A = (const float*)Ap;
#pragma unroll
      for (int it = 0; it < 4; it++) {
        int idx = it * 256 + t;             // 1024 chunks of 4 f32
        int row = idx >> 3, c4 = idx & 7;
        v4f v = *(const v4f*)(A + (size_t)(m0 + row) * DIM + kk + c4 * 4);
        v4bf o;
#pragma unroll
        for (int e = 0; e < 4; e++) o[e] = (bf16_t)v[e];
        *(v4bf*)&Abuf[buf][row * 32 + c4 * 4] = o;
      }
    }
  };
  auto loadB = [&](int kk, int buf) {
#pragma unroll
    for (int it = 0; it < 4; it++) {
      int idx = it * 256 + t;
      int row = idx >> 3, c4 = idx & 7;
      v4f v = *(const v4f*)(W + (size_t)(n0 + row) * DIM + kk + c4 * 4);
      v4bf o;
#pragma unroll
      for (int e = 0; e < 4; e++) o[e] = (bf16_t)v[e];
      *(v4bf*)&Bbuf[buf][row * 32 + c4 * 4] = o;
    }
  };

  loadA(0, 0);
  loadB(0, 0);
  __syncthreads();

  const int KSTEPS = DIM / 32;
  for (int kt = 0; kt < KSTEPS; kt++) {
    const int buf = kt & 1;
    if (kt + 1 < KSTEPS) { loadA((kt + 1) * 32, buf ^ 1); loadB((kt + 1) * 32, buf ^ 1); }
    v16bf aF[2], bF[4];
#pragma unroll
    for (int mi = 0; mi < 2; mi++) aF[mi] = frag_a(&Abuf[buf][(wm + mi * 16 + lm) * 32], h);
#pragma unroll
    for (int ni = 0; ni < 4; ni++) bF[ni] = frag_b(&Bbuf[buf][(wn + ni * 16 + lm) * 32], h);
#pragma unroll
    for (int mi = 0; mi < 2; mi++)
#pragma unroll
      for (int ni = 0; ni < 4; ni++) acc[mi][ni] = wmma_bf16(aF[mi], bF[ni], acc[mi][ni]);
    __syncthreads();
  }

#pragma unroll
  for (int mi = 0; mi < 2; mi++)
#pragma unroll
    for (int ni = 0; ni < 4; ni++)
#pragma unroll
      for (int r = 0; r < 8; r++) {
        const int grow = m0 + wm + mi * 16 + r + 8 * h;  // 0..4095
        const int gcol = n0 + wn + ni * 16 + lm;         // 0..2047
        const float val = acc[mi][ni][r];
        if (OMODE == 2) {
          ((float*)Out)[(size_t)grow * DIM + gcol] = val;
        } else {
          const int b = grow >> 11, s = grow & (SEQ - 1);
          const int hh = gcol >> 6, hd = gcol & (HD - 1);
          bf16_t* O = (bf16_t*)Out;
          if (OMODE == 0)
            O[((size_t)(b * HEADS + hh) * SEQ + s) * HD + hd] = (bf16_t)val;
          else
            O[((size_t)(b * HEADS + hh) * HD + hd) * SEQ + s] = (bf16_t)val;
        }
      }
}

// ---------------------------------------------------------------------------
// Sliding-window attention. Block = 128 threads (4 waves), one (b,h,64-query)
// tile per block; each wave owns 16 query rows; waves share K / V^T LDS tiles
// staged with CDNA5 async global->LDS copies (ASYNCcnt).
// Out-of-range keys: addresses are clamped into the valid sequence (finite
// data), and the mask is applied by SELECT so clamped rows never leak.
// ---------------------------------------------------------------------------
__global__ __launch_bounds__(128) void attn128(const bf16_t* __restrict__ Q,
                                               const bf16_t* __restrict__ K,
                                               const bf16_t* __restrict__ Vt,
                                               bf16_t* __restrict__ O) {
  __shared__ alignas(32) bf16_t kbuf[KT][HD];       // 40 KB
  __shared__ alignas(32) bf16_t vbuf[HD][KT];       // 40 KB (V transposed: [dim][key])
  __shared__ alignas(32) float  sbuf[4][16][KT];    // 80 KB raw scores per wave
  __shared__ float maxbuf[4][16];
  __shared__ float lbuf[4][16];

  const int t    = threadIdx.x;
  const int wave = t >> 5, lane = t & 31;
  const int lm = lane & 15, h = lane >> 4;
  const int bh = blockIdx.y;               // b*HEADS + h
  const int q0 = blockIdx.x * 64;
  const int kbase = q0 - WIN;              // key window [kbase, kbase+KT)
  const size_t headK = (size_t)bh * SEQ * HD;
  const size_t headV = (size_t)bh * HD * SEQ;

  // --- stage K tile (KT x 64 bf16) via async copies; clamp row into [0,SEQ) ---
  for (int it = 0; it < 20; it++) {
    int idx = it * 128 + t;                // 2560 chunks of 8 bf16
    int row = idx >> 3, c8 = idx & 7;
    int g = kbase + row;
    g = g < 0 ? 0 : (g >= SEQ ? SEQ - 1 : g);
    async_ld16(&kbuf[row][c8 * 8], K + headK + (size_t)g * HD + c8 * 8);
  }
  // --- stage V^T tile (64 x KT bf16); kbase is 64-aligned so 8-chunks don't straddle ---
  for (int it = 0; it < 20; it++) {
    int idx = it * 128 + t;
    int d = idx / 40, c = idx % 40;
    int g = kbase + c * 8;
    g = g < 0 ? 0 : (g > SEQ - 8 ? SEQ - 8 : g);
    async_ld16(&vbuf[d][c * 8], Vt + headV + (size_t)d * SEQ + g);
  }
  wait_async0();
  __syncthreads();

  // --- per-wave Q fragments (16 rows x 64 dims -> two K-chunks) ---
  const int qw = q0 + wave * 16;
  const bf16_t* qrow = Q + headK + (size_t)(qw + lm) * HD;
  const v16bf aQ0 = frag_a(qrow, h);
  const v16bf aQ1 = frag_a(qrow + 32, h);

  // --- pass 1: scores = select(mask, (Q K^T)/8, -1e30) -> LDS; track row max ---
  float vmax[8];
#pragma unroll
  for (int r = 0; r < 8; r++) vmax[r] = -3.0e38f;
  const int jl = kbase + lm;
  for (int jt = 0; jt < KT / 16; jt++) {
    v8f c = zero_v8f();
    c = wmma_bf16(aQ0, frag_b(&kbuf[jt * 16 + lm][0], h), c);
    c = wmma_bf16(aQ1, frag_b(&kbuf[jt * 16 + lm][32], h), c);
    const int j = jl + jt * 16;
    const bool jok = (j >= 0) && (j < SEQ);
#pragma unroll
    for (int r = 0; r < 8; r++) {
      const int q = qw + r + 8 * h;
      int d = q - j; if (d < 0) d = -d;
      const float s = (jok && d <= WIN) ? c[r] * 0.125f : -1.0e30f;  // 1/sqrt(64)
      sbuf[wave][r + 8 * h][jt * 16 + lm] = s;
      vmax[r] = fmaxf(vmax[r], s);
    }
  }
#pragma unroll
  for (int r = 0; r < 8; r++) {
    float v = vmax[r];
    v = fmaxf(v, __shfl_xor(v, 1));
    v = fmaxf(v, __shfl_xor(v, 2));
    v = fmaxf(v, __shfl_xor(v, 4));
    v = fmaxf(v, __shfl_xor(v, 8));
    vmax[r] = v;
  }
  if (lm == 0) {
#pragma unroll
    for (int r = 0; r < 8; r++) maxbuf[wave][r + 8 * h] = vmax[r];
  }
  __syncthreads();

  // --- pass 2: P = exp(S - max) as bf16 A-frags; O += P V ; accumulate row sums ---
  v8f o[4];
#pragma unroll
  for (int ni = 0; ni < 4; ni++) o[ni] = zero_v8f();
  float lsum = 0.f;
  const float rmax = maxbuf[wave][lm];
  const float* srow = sbuf[wave][lm];
  for (int kc = 0; kc < KT / 32; kc++) {
    v8f x0 = *(const v8f*)(srow + kc * 32 + 8 * h);
    v8f x1 = *(const v8f*)(srow + kc * 32 + 16 + 8 * h);
    v16bf aP;
#pragma unroll
    for (int e = 0; e < 8; e++) { float p = __expf(x0[e] - rmax); lsum += p; aP[e] = (bf16_t)p; }
#pragma unroll
    for (int e = 0; e < 8; e++) { float p = __expf(x1[e] - rmax); lsum += p; aP[8 + e] = (bf16_t)p; }
#pragma unroll
    for (int ni = 0; ni < 4; ni++)
      o[ni] = wmma_bf16(aP, frag_b(&vbuf[ni * 16 + lm][kc * 32], h), o[ni]);
  }
  lsum += __shfl_xor(lsum, 16);
  if (lane < 16) lbuf[wave][lm] = lsum;
  __syncthreads();

  float inv[8];
#pragma unroll
  for (int r = 0; r < 8; r++) inv[r] = 1.0f / lbuf[wave][r + 8 * h];

  const int b = bh >> 5, hh = bh & (HEADS - 1);
#pragma unroll
  for (int ni = 0; ni < 4; ni++)
#pragma unroll
    for (int r = 0; r < 8; r++) {
      const int q = qw + r + 8 * h;
      O[(size_t)(b * SEQ + q) * DIM + hh * HD + ni * 16 + lm] = (bf16_t)(o[ni][r] * inv[r]);
    }
}

// ---------------------------------------------------------------------------
extern "C" void kernel_launch(void* const* d_in, const int* in_sizes, int n_in,
                              void* d_out, int out_size, void* d_ws, size_t ws_size,
                              hipStream_t stream) {
  (void)in_sizes; (void)n_in; (void)out_size; (void)ws_size;
  const float* X  = (const float*)d_in[0];
  const float* Wq = (const float*)d_in[1];
  const float* Wk = (const float*)d_in[2];
  const float* Wv = (const float*)d_in[3];
  const float* Wo = (const float*)d_in[4];

  const size_t NBF = (size_t)BATCH * HEADS * SEQ * HD;  // 8M elements each
  bf16_t* Qb = (bf16_t*)d_ws;     // [B,H,S,Hd]
  bf16_t* Kb = Qb + NBF;          // [B,H,S,Hd]
  bf16_t* Vt = Kb + NBF;          // [B,H,Hd,S]
  bf16_t* AO = Vt + NBF;          // [B,S,D] attention output (bf16)

  dim3 grid(DIM / 128, MROWS / 128);  // (16, 32)
  gemm128<0, 0><<<grid, 256, 0, stream>>>(X, Wq, Qb);
  gemm128<0, 0><<<grid, 256, 0, stream>>>(X, Wk, Kb);
  gemm128<0, 1><<<grid, 256, 0, stream>>>(X, Wv, Vt);
  attn128<<<dim3(SEQ / 64, BATCH * HEADS), 128, 0, stream>>>(Qb, Kb, Vt, AO);
  gemm128<1, 2><<<grid, 256, 0, stream>>>(AO, Wo, (float*)d_out);
}
